// ProposalHead_62036507623503
// MI455X (gfx1250) — matrix operand order repaired and checked by reading, not compiled, not run on hardware
//
#include <hip/hip_runtime.h>
#include <hip/hip_bf16.h>

// ---------------------------------------------------------------------------
// Types / helpers
// ---------------------------------------------------------------------------
typedef __attribute__((ext_vector_type(16))) __bf16 v16bf;
typedef __attribute__((ext_vector_type(8)))  float  v8f;

union BF16V { v16bf v; unsigned short u[16]; };
union U128  { uint4 v; unsigned short u[8]; };
union U64V  { unsigned long long v; unsigned short u[4]; };

__device__ __forceinline__ unsigned short f2bf(float f) {
  unsigned u = __float_as_uint(f);
  u += 0x7FFFu + ((u >> 16) & 1u);          // round-to-nearest-even
  return (unsigned short)(u >> 16);
}
__device__ __forceinline__ float bf2f(unsigned short h) {
  return __uint_as_float(((unsigned)h) << 16);
}

// CDNA5 async global->LDS copy (ASYNCcnt tracked), one 128b packet per lane.
__device__ __forceinline__ void async_b128(unsigned lds_off,
                                           const unsigned short* g) {
  asm volatile("global_load_async_to_lds_b128 %0, %1, off"
               :: "v"(lds_off), "v"(g) : "memory");
}
__device__ __forceinline__ void wait_async0() {
  asm volatile("s_wait_asynccnt 0x0" ::: "memory");
}

__device__ __forceinline__ float block_sum256(float v, float* red) {
  const int tid = threadIdx.x, lane = tid & 31, wv = tid >> 5;
#pragma unroll
  for (int off = 16; off > 0; off >>= 1) v += __shfl_down(v, off, 32);
  if (lane == 0) red[wv] = v;
  __syncthreads();
  float r = 0.f;
  if (wv == 0) {
    r = (lane < (int)(blockDim.x >> 5)) ? red[lane] : 0.f;
#pragma unroll
    for (int off = 16; off > 0; off >>= 1) r += __shfl_down(r, off, 32);
    if (lane == 0) red[0] = r;
  }
  __syncthreads();
  r = red[0];
  __syncthreads();
  return r;
}

// ---------------------------------------------------------------------------
// Farthest point sampling: one block per batch, points resident in LDS
// ---------------------------------------------------------------------------
template <int NPTS>
__global__ void __launch_bounds__(512) fps_kernel(const float* __restrict__ xyz,
                                                  int npoint,
                                                  int* __restrict__ sel,
                                                  float* __restrict__ new_xyz) {
  constexpr int NTH = 512;
  constexpr int PPT = NPTS / NTH;
  __shared__ float sx[NPTS], sy[NPTS], sz[NPTS];
  __shared__ float swd[16];
  __shared__ int   swi[16];
  __shared__ float sfar[3];
  const int b = blockIdx.x;
  const int tid = threadIdx.x;
  const int lane = tid & 31, wv = tid >> 5;

  for (int i = tid; i < NPTS; i += NTH) {
    sx[i] = xyz[((size_t)b * NPTS + i) * 3 + 0];
    sy[i] = xyz[((size_t)b * NPTS + i) * 3 + 1];
    sz[i] = xyz[((size_t)b * NPTS + i) * 3 + 2];
  }
  float d[PPT];
#pragma unroll
  for (int p = 0; p < PPT; ++p) d[p] = 1e10f;
  __syncthreads();
  if (tid == 0) {
    sel[(size_t)b * npoint] = 0;
    new_xyz[((size_t)b * npoint) * 3 + 0] = sx[0];
    new_xyz[((size_t)b * npoint) * 3 + 1] = sy[0];
    new_xyz[((size_t)b * npoint) * 3 + 2] = sz[0];
    sfar[0] = sx[0]; sfar[1] = sy[0]; sfar[2] = sz[0];
  }
  __syncthreads();

  for (int it = 1; it < npoint; ++it) {
    const float fx = sfar[0], fy = sfar[1], fz = sfar[2];
    float bd = -1.f; int bi = 0x7fffffff;
#pragma unroll
    for (int p = 0; p < PPT; ++p) {
      const int i = tid + p * NTH;
      const float dx = sx[i] - fx, dy = sy[i] - fy, dz = sz[i] - fz;
      const float dd = dx * dx + dy * dy + dz * dz;
      d[p] = fminf(d[p], dd);
      if (d[p] > bd) { bd = d[p]; bi = i; }
    }
#pragma unroll
    for (int off = 16; off > 0; off >>= 1) {
      const float od = __shfl_down(bd, off, 32);
      const int   oi = __shfl_down(bi, off, 32);
      if (od > bd || (od == bd && oi < bi)) { bd = od; bi = oi; }
    }
    if (lane == 0) { swd[wv] = bd; swi[wv] = bi; }
    __syncthreads();
    if (wv == 0) {
      float rd = (lane < 16) ? swd[lane] : -2.f;
      int   ri = (lane < 16) ? swi[lane] : 0x7fffffff;
#pragma unroll
      for (int off = 8; off > 0; off >>= 1) {
        const float od = __shfl_down(rd, off, 32);
        const int   oi = __shfl_down(ri, off, 32);
        if (od > rd || (od == rd && oi < ri)) { rd = od; ri = oi; }
      }
      if (lane == 0) {
        sel[(size_t)b * npoint + it] = ri;
        new_xyz[((size_t)b * npoint + it) * 3 + 0] = sx[ri];
        new_xyz[((size_t)b * npoint + it) * 3 + 1] = sy[ri];
        new_xyz[((size_t)b * npoint + it) * 3 + 2] = sz[ri];
        sfar[0] = sx[ri]; sfar[1] = sy[ri]; sfar[2] = sz[ri];
      }
    }
    __syncthreads();
  }
}

// ---------------------------------------------------------------------------
// Ball query: first K indices (in index order) within radius, fill with first
// ---------------------------------------------------------------------------
__global__ void __launch_bounds__(512) ball_query_kernel(
    const float* __restrict__ xyz, const float* __restrict__ new_xyz,
    int Sin, int Sout, int K, float r2, int* __restrict__ gidx) {
  __shared__ float sp[4096 * 3];
  const int b = blockIdx.x;
  for (int i = threadIdx.x; i < Sin * 3; i += 512)
    sp[i] = xyz[(size_t)b * Sin * 3 + i];
  __syncthreads();
  for (int cen = threadIdx.x; cen < Sout; cen += 512) {
    const size_t base = (size_t)b * Sout + cen;
    const float cx = new_xyz[base * 3 + 0];
    const float cy = new_xyz[base * 3 + 1];
    const float cz = new_xyz[base * 3 + 2];
    int* out = &gidx[base * K];
    int cnt = 0, first = 0;
    for (int j = 0; j < Sin && cnt < K; ++j) {
      const float dx = sp[j * 3 + 0] - cx;
      const float dy = sp[j * 3 + 1] - cy;
      const float dz = sp[j * 3 + 2] - cz;
      if (dx * dx + dy * dy + dz * dz <= r2) {
        if (cnt == 0) first = j;
        out[cnt++] = j;
      }
    }
    for (; cnt < K; ++cnt) out[cnt] = first;
  }
}

// ---------------------------------------------------------------------------
// Gather grouped tensor (centered xyz | feats) as bf16, K-padded rows (ldap)
// ---------------------------------------------------------------------------
__global__ void __launch_bounds__(128) group_gather_kernel(
    const float* __restrict__ xyz, const unsigned short* __restrict__ feats,
    const float* __restrict__ new_xyz, const int* __restrict__ gidx,
    unsigned short* __restrict__ out, int Sin, int Sout, int K, int Cf,
    int ldap) {
  const int C = 3 + Cf;
  const int bs = blockIdx.x;              // b*Sout + s
  const int b = bs / Sout;
  __shared__ int   sidx[64];
  __shared__ float cc[3];
  if (threadIdx.x < K) sidx[threadIdx.x] = gidx[(size_t)bs * K + threadIdx.x];
  if (threadIdx.x < 3) cc[threadIdx.x] = new_xyz[(size_t)bs * 3 + threadIdx.x];
  __syncthreads();
  for (int e = threadIdx.x; e < K * ldap; e += blockDim.x) {
    const int k = e / ldap, c = e - k * ldap;
    const int j = sidx[k];
    unsigned short o = 0;
    if (c < 3)      o = f2bf(xyz[((size_t)b * Sin + j) * 3 + c] - cc[c]);
    else if (c < C) o = feats[((size_t)b * Sin + j) * Cf + (c - 3)];
    out[(size_t)bs * K * ldap + e] = o;
  }
}

// SA3: rows = [xyz2(3) | f2(Cf) | pad]
__global__ void concat3_kernel(const float* __restrict__ xyz2,
                               const unsigned short* __restrict__ f2,
                               unsigned short* __restrict__ out,
                               int rows, int Cf, int ldap) {
  const int C = 3 + Cf;
  const int total = rows * ldap;
  for (int i = blockIdx.x * blockDim.x + threadIdx.x; i < total;
       i += gridDim.x * blockDim.x) {
    const int r = i / ldap, c = i - r * ldap;
    unsigned short o = 0;
    if (c < 3)      o = f2bf(xyz2[(size_t)r * 3 + c]);
    else if (c < C) o = f2[(size_t)r * Cf + (c - 3)];
    out[i] = o;
  }
}

// Pre-pack layer weights: f32 (N,Kd) -> bf16 (N,ldap) zero-padded
__global__ void pack_weights_kernel(const float* __restrict__ W,
                                    unsigned short* __restrict__ Wb,
                                    int N, int Kd, int ldap) {
  const int i = blockIdx.x * blockDim.x + threadIdx.x;
  if (i < N * ldap) {
    const int n = i / ldap, k = i - n * ldap;
    Wb[i] = (k < Kd) ? f2bf(W[(size_t)n * Kd + k]) : (unsigned short)0;
  }
}

// ---------------------------------------------------------------------------
// WMMA GEMM:  H[M,N] = A[M,ldap](bf16) * Wb[N,ldap]^T(bf16), f32 accumulate.
// Tiles fetched with global_load_async_to_lds_b128, double-buffered LDS.
// Per-channel sum/sumsq accumulated into stats[2N] for BatchNorm.
// Block = 256 threads (8 waves). Tile 128(M) x 64(N), K-step 32.
// ---------------------------------------------------------------------------
__global__ void __launch_bounds__(256) gemm_bf16_stats(
    const unsigned short* __restrict__ A, const unsigned short* __restrict__ Wb,
    int lda, unsigned short* __restrict__ H, float* __restrict__ stats, int N) {
  __shared__ unsigned short As[2][128 * 40];   // stride 40: 16B-aligned b128
  __shared__ unsigned short Bs[2][64 * 40];
  __shared__ float s_sum[64], s_sq[64];

  const int tid = threadIdx.x;
  const int lane = tid & 31, wv = tid >> 5;
  const int r = lane & 15, g = lane >> 4;
  const int m0 = blockIdx.y * 128;
  const int n0 = blockIdx.x * 64;
  const int nsteps = lda >> 5;

  auto issue = [&](int k0, int buf) {
#pragma unroll
    for (int h = 0; h < 2; ++h) {              // A: 512 x 16B packets
      const int c = tid + h * 256;
      const int row = c >> 2, ko = (c & 3) * 8;
      async_b128((unsigned)(size_t)&As[buf][row * 40 + ko],
                 A + (size_t)(m0 + row) * lda + k0 + ko);
    }
    {                                          // W: 256 x 16B packets
      const int nn = tid >> 2, ko = (tid & 3) * 8;
      async_b128((unsigned)(size_t)&Bs[buf][nn * 40 + ko],
                 Wb + (size_t)(n0 + nn) * lda + k0 + ko);
    }
  };

  const v8f vzero = {0.f, 0.f, 0.f, 0.f, 0.f, 0.f, 0.f, 0.f};
  v8f acc[4] = {vzero, vzero, vzero, vzero};

  issue(0, 0);
  for (int s = 0; s < nsteps; ++s) {
    const int buf = s & 1;
    wait_async0();           // my packets for step s have landed in LDS
    __syncthreads();         // everyone's packets have landed
    if (s + 1 < nsteps) issue((s + 1) * 32, buf ^ 1);

    BF16V a;
    const unsigned short* ap = &As[buf][(wv * 16 + r) * 40 + g * 8];
#pragma unroll
    for (int j = 0; j < 8; ++j) { a.u[j] = ap[j]; a.u[j + 8] = ap[j + 16]; }
#pragma unroll
    for (int t = 0; t < 4; ++t) {
      BF16V bb;
      const unsigned short* bp = &Bs[buf][(t * 16 + r) * 40 + g * 8];
#pragma unroll
      for (int j = 0; j < 8; ++j) { bb.u[j] = bp[j]; bb.u[j + 8] = bp[j + 16]; }
      acc[t] = __builtin_amdgcn_wmma_f32_16x16x32_bf16(
          false, a.v, false, bb.v, (short)0, acc[t], false, false);
    }
    // no trailing barrier needed: ds reads are dscnt-waited before the WMMAs,
    // and the next async issue targets the other LDS buffer.
  }

  for (int i = tid; i < 64; i += 256) { s_sum[i] = 0.f; s_sq[i] = 0.f; }
  __syncthreads();
#pragma unroll
  for (int t = 0; t < 4; ++t) {
    float ls = 0.f, lq = 0.f;
#pragma unroll
    for (int v = 0; v < 8; ++v) {
      const float c = acc[t][v];
      ls += c; lq += c * c;
      H[(size_t)(m0 + wv * 16 + g * 8 + v) * N + (n0 + t * 16 + r)] = f2bf(c);
    }
    atomicAdd(&s_sum[t * 16 + r], ls);
    atomicAdd(&s_sq[t * 16 + r], lq);
  }
  __syncthreads();
  for (int i = tid; i < 64; i += 256) {
    atomicAdd(&stats[n0 + i], s_sum[i]);
    atomicAdd(&stats[N + n0 + i], s_sq[i]);
  }
}

// ---------------------------------------------------------------------------
// BN finalize: scale = g*rsqrt(var+eps), shift = beta - mu*scale
// (conv bias cancels exactly under global BatchNorm, so it is omitted)
// ---------------------------------------------------------------------------
__global__ void bn_finalize_kernel(const float* __restrict__ stats,
                                   const float* __restrict__ g,
                                   const float* __restrict__ beta,
                                   float* __restrict__ scale,
                                   float* __restrict__ shift,
                                   int N, float inv_count) {
  const int c = blockIdx.x * blockDim.x + threadIdx.x;
  if (c < N) {
    const float mu = stats[c] * inv_count;
    const float var = stats[N + c] * inv_count - mu * mu;
    const float s = g[c] * rsqrtf(var + 1e-5f);
    scale[c] = s;
    shift[c] = beta[c] - mu * s;
  }
}

// elementwise BN affine + ReLU, 8 bf16 per thread (128-bit loads/stores)
__global__ void bn_relu_kernel(const unsigned short* __restrict__ H,
                               unsigned short* __restrict__ out,
                               const float* __restrict__ scale,
                               const float* __restrict__ shift,
                               size_t total8, int N) {
  const size_t i = (size_t)blockIdx.x * blockDim.x + threadIdx.x;
  if (i >= total8) return;
  const size_t base = i * 8;
  const int c0 = (int)(base % (size_t)N);   // N % 8 == 0: no row crossing
  U128 h; h.v = *(const uint4*)(H + base);
  U128 o;
#pragma unroll
  for (int j = 0; j < 8; ++j) {
    const float v = bf2f(h.u[j]) * scale[c0 + j] + shift[c0 + j];
    o.u[j] = f2bf(fmaxf(v, 0.f));
  }
  *(uint4*)(out + base) = o.v;
}

// BN affine + ReLU + max over K, 4 channels per thread
template <bool OUTF32>
__global__ void __launch_bounds__(128) bn_relu_max_kernel(
    const unsigned short* __restrict__ H, void* __restrict__ outv,
    const float* __restrict__ scale, const float* __restrict__ shift,
    int K, int N) {
  const int bs = blockIdx.x;
  const int c0 = (blockIdx.y * blockDim.x + threadIdx.x) * 4;
  if (c0 >= N) return;
  float sc[4], sh[4], m[4];
#pragma unroll
  for (int j = 0; j < 4; ++j) { sc[j] = scale[c0 + j]; sh[j] = shift[c0 + j]; m[j] = 0.f; }
  const unsigned short* p = H + (size_t)bs * K * N + c0;
  for (int k = 0; k < K; ++k) {
    U64V h; h.v = *(const unsigned long long*)(p + (size_t)k * N);
#pragma unroll
    for (int j = 0; j < 4; ++j) m[j] = fmaxf(m[j], bf2f(h.u[j]) * sc[j] + sh[j]);
  }
  if (OUTF32) {
#pragma unroll
    for (int j = 0; j < 4; ++j) ((float*)outv)[(size_t)bs * N + c0 + j] = m[j];
  } else {
    U64V o;
#pragma unroll
    for (int j = 0; j < 4; ++j) o.u[j] = f2bf(m[j]);
    *(unsigned long long*)((unsigned short*)outv + (size_t)bs * N + c0) = o.v;
  }
}

// ---------------------------------------------------------------------------
// FPN projection: h = ctx @ Wp^T + bp ; per-row LayerNorm ; exact GELU
// ---------------------------------------------------------------------------
__global__ void __launch_bounds__(256) fpn_kernel(
    const float* __restrict__ ctx, const float* __restrict__ W,
    const float* __restrict__ bias, const float* __restrict__ lng,
    const float* __restrict__ lnb, float* __restrict__ out) {
  __shared__ float sctx[256];
  __shared__ float red[32];
  const int b = blockIdx.x, tid = threadIdx.x;
  sctx[tid] = ctx[b * 256 + tid];
  __syncthreads();
  float h[4];
#pragma unroll
  for (int q = 0; q < 4; ++q) {
    const int c = tid + q * 256;
    float a = bias[c];
    const float* wr = &W[(size_t)c * 256];
    for (int k = 0; k < 256; ++k) a += sctx[k] * wr[k];
    h[q] = a;
  }
  const float ls = h[0] + h[1] + h[2] + h[3];
  const float lq = h[0]*h[0] + h[1]*h[1] + h[2]*h[2] + h[3]*h[3];
  const float S = block_sum256(ls, red);
  const float Q = block_sum256(lq, red);
  const float mu = S * (1.f / 1024.f);
  const float var = Q * (1.f / 1024.f) - mu * mu;
  const float rs = rsqrtf(var + 1e-5f);
#pragma unroll
  for (int q = 0; q < 4; ++q) {
    const int c = tid + q * 256;
    const float v = (h[q] - mu) * rs * lng[c] + lnb[c];
    out[b * 1024 + c] = 0.5f * v * (1.f + erff(v * 0.70710678118654752f));
  }
}

// ---------------------------------------------------------------------------
// Fusion + cls/reg heads, one block per batch row
// ---------------------------------------------------------------------------
__global__ void __launch_bounds__(256) fusion_head_kernel(
    const float* __restrict__ feat, const float* __restrict__ fpn,
    const float* __restrict__ Wf, const float* __restrict__ bf,
    const float* __restrict__ Wc, const float* __restrict__ bc,
    const float* __restrict__ Wr, const float* __restrict__ br,
    float* __restrict__ out) {
  __shared__ float scat[2048];
  __shared__ float sfused[1024];
  __shared__ float red[32];
  const int b = blockIdx.x, tid = threadIdx.x;
  for (int i = tid; i < 1024; i += 256) {
    scat[i] = feat[b * 1024 + i];
    scat[1024 + i] = fpn[b * 1024 + i];
  }
  __syncthreads();
#pragma unroll
  for (int q = 0; q < 4; ++q) {
    const int c = tid + q * 256;
    float a = bf[c];
    const float* wr = &Wf[(size_t)c * 2048];
    for (int k = 0; k < 2048; ++k) a += scat[k] * wr[k];
    sfused[c] = a;
  }
  __syncthreads();
  for (int j = 0; j < 7; ++j) {
    const float* wrow = (j == 0) ? Wc : (Wr + (size_t)(j - 1) * 1024);
    float p = 0.f;
    for (int k = tid; k < 1024; k += 256) p += sfused[k] * wrow[k];
    const float tot = block_sum256(p, red);
    if (tid == 0) {
      if (j == 0) out[b] = tot + bc[0];
      else out[16 + b * 6 + (j - 1)] = tot + br[j - 1];
    }
  }
}

// ---------------------------------------------------------------------------
// Host orchestration
// ---------------------------------------------------------------------------
extern "C" void kernel_launch(void* const* d_in, const int* in_sizes, int n_in,
                              void* d_out, int out_size, void* d_ws, size_t ws_size,
                              hipStream_t stream) {
  (void)in_sizes; (void)n_in; (void)out_size; (void)ws_size;
  const float* xyz = (const float*)d_in[0];
  const float* ctx = (const float*)d_in[1];
  const float *W_sa[9], *g_sa[9], *be_sa[9];
  for (int l = 0; l < 9; ++l) {
    W_sa[l]  = (const float*)d_in[2 + l * 4 + 0];
    g_sa[l]  = (const float*)d_in[2 + l * 4 + 2];
    be_sa[l] = (const float*)d_in[2 + l * 4 + 3];
  }
  const float* Wp  = (const float*)d_in[38];
  const float* bp  = (const float*)d_in[39];
  const float* lng = (const float*)d_in[40];
  const float* lnb = (const float*)d_in[41];
  const float* Wf  = (const float*)d_in[42];
  const float* bfv = (const float*)d_in[43];
  const float* Wc  = (const float*)d_in[44];
  const float* bc  = (const float*)d_in[45];
  const float* Wr  = (const float*)d_in[46];
  const float* br  = (const float*)d_in[47];

  char* ws = (char*)d_ws;
  size_t off = 0;
  auto alloc = [&](size_t bytes) -> void* {
    void* p = ws + off;
    off = (off + bytes + 255) & ~(size_t)255;
    return p;
  };
  unsigned short* bufA = (unsigned short*)alloc((size_t)262144 * 160 * 2); // 83.9 MB
  unsigned short* bufB = (unsigned short*)alloc((size_t)524288 * 128 * 2); // 134 MB
  unsigned short* Wb   = (unsigned short*)alloc((size_t)1024 * 512 * 2);   // packed weights
  int*   sel1  = (int*)alloc((size_t)16 * 1024 * 4);
  float* nxyz1 = (float*)alloc((size_t)16 * 1024 * 3 * 4);
  int*   gidx1 = (int*)alloc((size_t)16 * 1024 * 32 * 4);
  unsigned short* f1 = (unsigned short*)alloc((size_t)16 * 1024 * 128 * 2);
  int*   sel2  = (int*)alloc((size_t)16 * 256 * 4);
  float* nxyz2 = (float*)alloc((size_t)16 * 256 * 3 * 4);
  int*   gidx2 = (int*)alloc((size_t)16 * 256 * 64 * 4);
  unsigned short* f2 = (unsigned short*)alloc((size_t)16 * 256 * 256 * 2);
  float* feat  = (float*)alloc((size_t)16 * 1024 * 4);
  float* fpnf  = (float*)alloc((size_t)16 * 1024 * 4);
  float* stats = (float*)alloc((size_t)2 * 1024 * 4);
  float* scale = (float*)alloc((size_t)1024 * 4);
  float* shift = (float*)alloc((size_t)1024 * 4);

  auto run_layer = [&](const unsigned short* Ain, int ldap, unsigned short* Hout,
                       int M, int Kd, int N, int li) {
    pack_weights_kernel<<<(N * ldap + 255) / 256, 256, 0, stream>>>(
        W_sa[li], Wb, N, Kd, ldap);
    hipMemsetAsync(stats, 0, (size_t)2 * N * sizeof(float), stream);
    dim3 grid(N / 64, M / 128);
    gemm_bf16_stats<<<grid, 256, 0, stream>>>(Ain, Wb, ldap, Hout, stats, N);
    bn_finalize_kernel<<<(N + 255) / 256, 256, 0, stream>>>(
        stats, g_sa[li], be_sa[li], scale, shift, N, 1.f / (float)M);
  };
  auto relu_pass = [&](const unsigned short* Hin, unsigned short* Aout,
                       size_t total, int N) {
    const size_t t8 = total / 8;
    bn_relu_kernel<<<(unsigned)((t8 + 255) / 256), 256, 0, stream>>>(
        Hin, Aout, scale, shift, t8, N);
  };

  // ---------------- SA1 ----------------
  fps_kernel<4096><<<16, 512, 0, stream>>>(xyz, 1024, sel1, nxyz1);
  ball_query_kernel<<<16, 512, 0, stream>>>(xyz, nxyz1, 4096, 1024, 32, 0.04f, gidx1);
  group_gather_kernel<<<16 * 1024, 128, 0, stream>>>(
      xyz, (const unsigned short*)nullptr, nxyz1, gidx1, bufA, 4096, 1024, 32, 0, 32);
  const int M1 = 16 * 1024 * 32;
  run_layer(bufA, 32, bufB, M1, 3, 64, 0);
  relu_pass(bufB, bufA, (size_t)M1 * 64, 64);
  run_layer(bufA, 64, bufB, M1, 64, 64, 1);
  relu_pass(bufB, bufA, (size_t)M1 * 64, 64);
  run_layer(bufA, 64, bufB, M1, 64, 128, 2);
  bn_relu_max_kernel<false><<<dim3(16 * 1024, 1), 128, 0, stream>>>(
      bufB, (void*)f1, scale, shift, 32, 128);

  // ---------------- SA2 ----------------
  fps_kernel<1024><<<16, 512, 0, stream>>>(nxyz1, 256, sel2, nxyz2);
  ball_query_kernel<<<16, 512, 0, stream>>>(nxyz1, nxyz2, 1024, 256, 64, 0.16f, gidx2);
  group_gather_kernel<<<16 * 256, 128, 0, stream>>>(
      nxyz1, f1, nxyz2, gidx2, bufA, 1024, 256, 64, 128, 160);
  const int M2 = 16 * 256 * 64;
  run_layer(bufA, 160, bufB, M2, 131, 128, 3);
  relu_pass(bufB, bufA, (size_t)M2 * 128, 128);
  run_layer(bufA, 128, bufB, M2, 128, 128, 4);
  relu_pass(bufB, bufA, (size_t)M2 * 128, 128);
  run_layer(bufA, 128, bufB, M2, 128, 256, 5);
  bn_relu_max_kernel<false><<<dim3(16 * 256, 1), 128, 0, stream>>>(
      bufB, (void*)f2, scale, shift, 64, 256);

  // ---------------- SA3 (global) ----------------
  concat3_kernel<<<(4096 * 288 + 255) / 256, 256, 0, stream>>>(
      nxyz2, f2, bufA, 4096, 256, 288);
  const int M3 = 4096;
  run_layer(bufA, 288, bufB, M3, 259, 256, 6);
  relu_pass(bufB, bufA, (size_t)M3 * 256, 256);
  run_layer(bufA, 256, bufB, M3, 256, 512, 7);
  relu_pass(bufB, bufA, (size_t)M3 * 512, 512);
  run_layer(bufA, 512, bufB, M3, 512, 1024, 8);
  bn_relu_max_kernel<true><<<dim3(16, 2), 128, 0, stream>>>(
      bufB, (void*)feat, scale, shift, 256, 1024);

  // ---------------- Heads ----------------
  fpn_kernel<<<16, 256, 0, stream>>>(ctx, Wp, bp, lng, lnb, fpnf);
  fusion_head_kernel<<<16, 256, 0, stream>>>(feat, fpnf, Wf, bfv, Wc, bc, Wr, br,
                                             (float*)d_out);
}
